// GMFlowMatching_18777597018136
// MI455X (gfx1250) — compile-verified, matrix operand order checked by reading, not run on hardware
//
#include <hip/hip_runtime.h>
#include <hip/hip_fp16.h>

typedef __attribute__((ext_vector_type(16))) _Float16 v16h;
typedef __attribute__((ext_vector_type(8)))  float    v8f;

#define B_  2
#define C_  128
#define H_  96
#define W_  96
#define HW_ (H_ * W_)            // 9216
#define NSTRIPE (HW_ / 64)       // 144 stripes of 64 columns
#define KROW 136                 // padded LDS row stride in f16 (272B, 16B aligned)

// Fold (1/sqrt(128))/TEMP and log2(e) into the query pre-scale so the WMMA
// accumulator is directly in the log2 domain: e = exp2(acc) == exp(corr/T).
#define PRE_SCALE 1.2751743832351615f   // 0.8838834764831844 * 1.4426950408889634
#define CLAMP2    72.13475204444817f    // 50 * log2(e)

union Frag16 { v16h h; int4 q[2]; };

// ---------------------------------------------------------------------------
// Kernel 1: per-pixel inverse L2 norm over channels (coalesced across pixels)
// ---------------------------------------------------------------------------
__global__ __launch_bounds__(256) void gm_norm_kernel(const float* __restrict__ fL,
                                                      const float* __restrict__ fR,
                                                      float* __restrict__ rnL,
                                                      float* __restrict__ rnR) {
  int idx = blockIdx.x * 256 + threadIdx.x;      // B*HW threads
  int b = idx / HW_;
  int p = idx - b * HW_;
  const size_t base = (size_t)b * C_ * HW_ + p;
  float sL = 0.f, sR = 0.f;
#pragma unroll 4
  for (int c = 0; c < C_; ++c) {
    float a = fL[base + (size_t)c * HW_];
    float r = fR[base + (size_t)c * HW_];
    sL = fmaf(a, a, sL);
    sR = fmaf(r, r, sR);
  }
  rnL[idx] = 1.f / fmaxf(sqrtf(sL), 1e-6f);
  rnR[idx] = 1.f / fmaxf(sqrtf(sR), 1e-6f);
}

// ---------------------------------------------------------------------------
// Kernel 2: normalize (+optional pre-scale) + transpose [C,HW] f32 -> [HW,C] f16
// ---------------------------------------------------------------------------
__global__ __launch_bounds__(256) void gm_pack_kernel(const float* __restrict__ f,
                                                      const float* __restrict__ rn,
                                                      _Float16* __restrict__ o,
                                                      float sc) {
  __shared__ __align__(16) _Float16 tile[32 * KROW];
  const int tid = threadIdx.x;
  const int b = blockIdx.x / (HW_ / 32);         // 288 tiles per batch
  const int pbase = (blockIdx.x % (HW_ / 32)) * 32;
  const int px = tid & 31;
  const int cg = tid >> 5;                       // 8 channel groups
  const float rv = rn[b * HW_ + pbase + px] * sc;
  const size_t inb = (size_t)b * C_ * HW_ + pbase + px;
#pragma unroll
  for (int j = 0; j < 16; ++j) {
    int c = cg * 16 + j;
    float v = f[inb + (size_t)c * HW_];
    tile[px * KROW + c] = (_Float16)(v * rv);
  }
  __syncthreads();
  // write 32B per thread, contiguous in channel-major output
  const int p2 = tid >> 3;                       // 0..31
  const int cs = (tid & 7) * 16;                 // 0..112
  const int4* src = (const int4*)&tile[p2 * KROW + cs];
  int4* dst = (int4*)&o[((size_t)b * HW_ + pbase + p2) * C_ + cs];
  dst[0] = src[0];
  dst[1] = src[1];
}

// ---------------------------------------------------------------------------
// Kernel 3: fused WMMA correlation + streaming softmax + flow/conf output.
// Block = 256 threads = 8 waves; each wave owns 16 query rows.
// Columns streamed in 64-wide stripes, async-copied to double-buffered LDS.
// ---------------------------------------------------------------------------
__global__ __launch_bounds__(256, 1) void gm_main_kernel(const _Float16* __restrict__ qn,
                                                         const _Float16* __restrict__ kn,
                                                         float* __restrict__ out) {
  __shared__ __align__(16) _Float16 ktile[2][64 * KROW];   // 2 x 17408 B

  const int tid  = threadIdx.x;
  const int lane = tid & 31;
  const int wid  = tid >> 5;
  const int b       = blockIdx.x / (HW_ / 128);  // 72 row-blocks per batch
  const int rowbase = (blockIdx.x % (HW_ / 128)) * 128 + wid * 16;
  const int m   = lane & 15;
  const int hi8 = (lane >> 4) * 8;               // K sub-offset AND D-row offset

  // ---- A fragments (16 rows x 128 ch), resident in VGPRs ----
  Frag16 afr[4];
  const _Float16* qrow = qn + ((size_t)b * HW_ + rowbase + m) * C_;
#pragma unroll
  for (int kc = 0; kc < 4; ++kc) {
    afr[kc].q[0] = *(const int4*)(qrow + kc * 32 + hi8);
    afr[kc].q[1] = *(const int4*)(qrow + kc * 32 + 16 + hi8);
  }

  float l[8], sx[8], sy[8], em[8];
#pragma unroll
  for (int j = 0; j < 8; ++j) { l[j] = 0.f; sx[j] = 0.f; sy[j] = 0.f; em[j] = 0.f; }

  const unsigned long long knbase =
      (unsigned long long)(uintptr_t)kn + (unsigned long long)b * ((size_t)HW_ * C_ * 2);
  const unsigned lds0 = (unsigned)(uintptr_t)&ktile[0][0];
  const unsigned lds1 = (unsigned)(uintptr_t)&ktile[1][0];
  // thread -> (pixel-in-stripe, 16B part): 4 chunks per thread
  const int ppix  = tid >> 2;                    // 0..63
  const int ppart = (tid & 3) * 4;               // 0,4,8,12

  // ---- prefetch stripe 0 ----
  {
    unsigned la = lds0 + (unsigned)(ppix * (KROW * 2) + ppart * 16);
    unsigned long long ga = knbase + (unsigned long long)(ppix * 256 + ppart * 16);
#pragma unroll
    for (int u = 0; u < 4; ++u)
      asm volatile("global_load_async_to_lds_b128 %0, %1, off"
                   :: "v"(la + u * 16u), "v"(ga + (unsigned long long)(u * 16)) : "memory");
  }

  for (int i = 0; i < NSTRIPE; ++i) {
    asm volatile("s_wait_asynccnt 0" ::: "memory");
    __syncthreads();                              // stripe i resident for all waves
    if (i + 1 < NSTRIPE) {                        // prefetch stripe i+1 into other buffer
      unsigned la = (((i + 1) & 1) ? lds1 : lds0) + (unsigned)(ppix * (KROW * 2) + ppart * 16);
      unsigned long long ga = knbase + (unsigned long long)(i + 1) * (64ull * C_ * 2)
                            + (unsigned long long)(ppix * 256 + ppart * 16);
#pragma unroll
      for (int u = 0; u < 4; ++u)
        asm volatile("global_load_async_to_lds_b128 %0, %1, off"
                     :: "v"(la + u * 16u), "v"(ga + (unsigned long long)(u * 16)) : "memory");
    }

    const _Float16* kt = &ktile[i & 1][0];
#pragma unroll
    for (int t4 = 0; t4 < 4; ++t4) {
      const _Float16* kp = kt + (t4 * 16 + m) * KROW + hi8;
      v8f acc = {};
#pragma unroll
      for (int kc = 0; kc < 4; ++kc) {
        Frag16 bfr;
        bfr.q[0] = *(const int4*)(kp + kc * 32);
        bfr.q[1] = *(const int4*)(kp + kc * 32 + 16);
        acc = __builtin_amdgcn_wmma_f32_16x16x32_f16(
            false, afr[kc].h, false, bfr.h, (short)0, acc, false, false);
      }
      const int colpix = i * 64 + t4 * 16 + m;
      const float cx = (float)(colpix % W_);
      const float cy = (float)(colpix / W_);
#pragma unroll
      for (int j = 0; j < 8; ++j) {
        // acc is already (corr/T)*log2e thanks to the folded query pre-scale.
        float z = __builtin_amdgcn_fmed3f(acc[j], -CLAMP2, CLAMP2); // v_med3_num_f32
        float e = __builtin_amdgcn_exp2f(z);                        // bare v_exp_f32
        em[j] = fmaxf(em[j], e);
        l[j] += e;
        sx[j] = fmaf(e, cx, sx[j]);
        sy[j] = fmaf(e, cy, sy[j]);
      }
    }
  }

  // ---- merge the 16 column-lanes per half-wave (masks stay inside halves) ----
#pragma unroll
  for (int j = 0; j < 8; ++j) {
#pragma unroll
    for (int msk = 1; msk < 16; msk <<= 1) {
      l[j]  += __shfl_xor(l[j],  msk, 32);
      sx[j] += __shfl_xor(sx[j], msk, 32);
      sy[j] += __shfl_xor(sy[j], msk, 32);
      em[j]  = fmaxf(em[j], __shfl_xor(em[j], msk, 32));
    }
  }

  if (m == 0) {                                   // lane 0 -> rows 0..7, lane 16 -> rows 8..15
    const int r0 = rowbase + hi8;
#pragma unroll
    for (int j = 0; j < 8; ++j) {
      int row = r0 + j;                           // pixel index within batch
      float inv = 1.f / l[j];
      float fx = sx[j] * inv - (float)(row % W_);
      float fy = sy[j] * inv - (float)(row / W_);
      out[(size_t)b * 2 * HW_ + row]        = fx;               // flow[b][0]
      out[(size_t)b * 2 * HW_ + HW_ + row]  = fy;               // flow[b][1]
      out[(size_t)B_ * 2 * HW_ + (size_t)b * HW_ + row] = em[j] * inv;  // conf
    }
  }
}

// ---------------------------------------------------------------------------
extern "C" void kernel_launch(void* const* d_in, const int* in_sizes, int n_in,
                              void* d_out, int out_size, void* d_ws, size_t ws_size,
                              hipStream_t stream) {
  (void)in_sizes; (void)n_in; (void)out_size; (void)ws_size;
  const float* fL = (const float*)d_in[0];
  const float* fR = (const float*)d_in[1];
  float* out = (float*)d_out;

  char* ws = (char*)d_ws;
  float* rnL = (float*)ws;                                   // B*HW f32
  float* rnR = rnL + (size_t)B_ * HW_;
  _Float16* qn = (_Float16*)(ws + 2u * B_ * HW_ * sizeof(float));   // 147456 B offset (256-aligned)
  _Float16* kn = qn + (size_t)B_ * HW_ * C_;                        // +4718592 B (256-aligned)

  gm_norm_kernel<<<(B_ * HW_) / 256, 256, 0, stream>>>(fL, fR, rnL, rnR);
  gm_pack_kernel<<<B_ * (HW_ / 32), 256, 0, stream>>>(fL, rnL, qn, PRE_SCALE);
  gm_pack_kernel<<<B_ * (HW_ / 32), 256, 0, stream>>>(fR, rnR, kn, 1.0f);
  gm_main_kernel<<<B_ * (HW_ / 128), 256, 0, stream>>>(qn, kn, out);
}
